// TemporalPropagator_55276229099699
// MI455X (gfx1250) — matrix operand order repaired and checked by reading, not compiled, not run on hardware
//
#include <hip/hip_runtime.h>
#include <stdint.h>

#define DD 1024
#define BB 16
#define SPP 256
#define TT (BB*SPP)   /* 4096 */
#define HH (DD/4)     /* 256  */
#define NTERMS 16

typedef __attribute__((ext_vector_type(16))) __bf16 v16bf;
typedef __attribute__((ext_vector_type(8)))  float  v8f;
typedef __attribute__((ext_vector_type(4)))  uint32_t v4u;

union Frag { uint32_t u[8]; v16bf v; };

__device__ __forceinline__ uint16_t f2bf(float x) {
  uint32_t u = __builtin_bit_cast(uint32_t, x);
  u += 0x7FFFu + ((u >> 16) & 1u);          // round-to-nearest-even
  return (uint16_t)(u >> 16);
}

// ---- Fragment loaders per CDNA5 ISA VGPR layouts (wave32) ----

// A matrix 16x32 bf16, row-major source, lda in elements.
__device__ __forceinline__ Frag load_a_frag(const uint16_t* base, int lda, int row0, int k0, int lane) {
  Frag f;
  const int m = lane & 15, half = lane >> 4;
  const uint16_t* p = base + (size_t)(row0 + m) * lda + k0;
#pragma unroll
  for (int j = 0; j < 4; ++j)
    f.u[j] = *(const uint32_t*)(p + 8*half + 2*j);
#pragma unroll
  for (int j = 0; j < 4; ++j)
    f.u[4+j] = *(const uint32_t*)(p + 16 + 8*half + 2*j);
  return f;
}

// B matrix 32x16 with the operand stored transposed: N x K row-major ("nt").
__device__ __forceinline__ Frag load_b_nt(const uint16_t* base, int ldb, int col0, int k0, int lane) {
  Frag f;
  const int n = lane & 15, half = lane >> 4;
  const uint16_t* p = base + (size_t)(col0 + n) * ldb + k0;
#pragma unroll
  for (int j = 0; j < 4; ++j)
    f.u[j] = *(const uint32_t*)(p + 8*half + 2*j);
#pragma unroll
  for (int j = 0; j < 4; ++j)
    f.u[4+j] = *(const uint32_t*)(p + 16 + 8*half + 2*j);
  return f;
}

__device__ __forceinline__ v8f wmma_bf(const Frag& a, const Frag& b, v8f c) {
  return __builtin_amdgcn_wmma_f32_16x16x32_bf16(false, a.v, false, b.v, (short)0, c, false, false);
}

// ---------------- utility kernels ----------------

__global__ void k_f32_to_bf16(const float* __restrict__ in, uint16_t* __restrict__ out, int n) {
  int i = blockIdx.x * blockDim.x + threadIdx.x;
  if (i < n) out[i] = f2bf(in[i]);
}

// P = A^T with A = (vre - vre^T) + i(vim + vim^T).
// P_re antisymmetric, P_im symmetric -> P^T comes free:  Pt_re = -P_re, Pt_im = P_im.
// Vt init = I - 2P (k=1 Neumann term of V^T = I + 2*sum_{k>=1} (-1)^k P^k).
__global__ void k_prep(const float* __restrict__ vre, const float* __restrict__ vim,
                       uint16_t* __restrict__ Pre, uint16_t* __restrict__ Pim,
                       uint16_t* __restrict__ PtR, uint16_t* __restrict__ PtI,
                       float* __restrict__ Vre, float* __restrict__ Vim) {
  int idx = blockIdx.x * blockDim.x + threadIdx.x;
  if (idx >= DD*DD) return;
  int i = idx / DD, j = idx % DD;
  float pr = vre[(size_t)j*DD + i] - vre[(size_t)i*DD + j];
  float pi = vim[(size_t)j*DD + i] + vim[(size_t)i*DD + j];
  Pre[idx] = f2bf(pr);  Pim[idx] = f2bf(pi);
  PtR[idx] = f2bf(-pr); PtI[idx] = f2bf(pi);
  Vre[idx] = (i == j ? 1.0f : 0.0f) - 2.0f * pr;
  Vim[idx] = -2.0f * pi;
}

// M_out = P * M_in (complex DxD); M_in/M_out kept TRANSPOSED (NxK) so the B
// operand is two contiguous b128 loads per fragment.  Vt += sign2 * M_out.
__global__ void k_neumann(const uint16_t* __restrict__ Pre, const uint16_t* __restrict__ Pim,
                          const uint16_t* __restrict__ MtR, const uint16_t* __restrict__ MtI,
                          uint16_t* __restrict__ OtR, uint16_t* __restrict__ OtI,
                          float* __restrict__ Vre, float* __restrict__ Vim, float sign2) {
  const int lane = threadIdx.x & 31;
  const int wid  = blockIdx.x * (blockDim.x >> 5) + (threadIdx.x >> 5);
  const int tn = wid & 63, tm = wid >> 6;            // 64x64 tiles
  const int row0 = tm * 16, col0 = tn * 16;
  v8f cr = {}; v8f ci = {};
  for (int k0 = 0; k0 < DD; k0 += 32) {
    Frag are = load_a_frag(Pre, DD, row0, k0, lane);
    Frag aim = load_a_frag(Pim, DD, row0, k0, lane);
    Frag ain;
#pragma unroll
    for (int j = 0; j < 8; ++j) ain.u[j] = aim.u[j] ^ 0x80008000u;
    Frag bre = load_b_nt(MtR, DD, col0, k0, lane);
    Frag bim = load_b_nt(MtI, DD, col0, k0, lane);
    cr = wmma_bf(are, bre, cr);
    cr = wmma_bf(ain, bim, cr);   // - Aim*Bim
    ci = wmma_bf(are, bim, ci);
    ci = wmma_bf(aim, bre, ci);
  }
  const int n = col0 + (lane & 15), half = lane >> 4;
#pragma unroll
  for (int r = 0; r < 8; ++r) {
    int m = row0 + r + 8*half;
    size_t idn = (size_t)m * DD + n;        // natural (for f32 accumulator)
    Vre[idn] += sign2 * cr[r]; Vim[idn] += sign2 * ci[r];
    size_t idt = (size_t)n * DD + m;        // transposed bf16 (contiguous per lane)
    OtR[idt] = f2bf(cr[r]);  OtI[idt] = f2bf(ci[r]);
  }
}

// Wt[n,k] = Vt[n,k]*s[n]*(1-a) + DFT[n,k]*a   (DFT symmetric; 1/sqrt(1024)=1/32)
__global__ void k_combine(const float* __restrict__ Vre, const float* __restrict__ Vim,
                          const float* __restrict__ log_sigma, const float* __restrict__ dftw,
                          uint16_t* __restrict__ Wtr, uint16_t* __restrict__ Wti) {
  int idx = blockIdx.x * blockDim.x + threadIdx.x;
  if (idx >= DD*DD) return;
  int n = idx / DD, k = idx % DD;
  float a = 1.0f / (1.0f + __expf(-dftw[0]));
  float s = __expf(log_sigma[n]);
  float th = -6.28318530718f * (float)((n * k) & (DD - 1)) / (float)DD;
  float wr = Vre[idx] * s * (1.0f - a) + __cosf(th) * 0.03125f * a;
  float wi = Vim[idx] * s * (1.0f - a) + __sinf(th) * 0.03125f * a;
  Wtr[idx] = f2bf(wr); Wti[idx] = f2bf(wi);
}

// x_tilde = x @ W (x real): the 16x32 A tile is shared by all 4 waves of the
// block, so stage it once per k-slab through the async global->LDS path
// (double buffered, ASYNCcnt + barrier), then read A fragments with ds_load.
__global__ void k_xtilde(const uint16_t* __restrict__ X, const uint16_t* __restrict__ Wtr,
                         const uint16_t* __restrict__ Wti,
                         float* __restrict__ outR, float* __restrict__ outI) {
  __shared__ __align__(16) uint32_t sh32[512];        // 2 x (16 rows x 64B)
  const int tid  = threadIdx.x;
  const int lane = tid & 31;
  const int wid  = blockIdx.x * 4 + (tid >> 5);
  const int tn = wid & 63, tm = wid >> 6;             // (T/16)=256 x (D/16)=64
  const int row0 = tm * 16, col0 = tn * 16;

  // Escape the LDS array's address into the asm (ptrtoint capture) so the
  // allocation stays live and the "memory" clobber covers it.
  const uint64_t shesc = (uint64_t)(uintptr_t)sh32;

  // threads 0..63 (waves 0 and 1, fully active) move the A tile: 16 rows x 4 x 16B
  const int srow = tid >> 2, schk = tid & 3;
  const uint16_t* gsrc = X + (size_t)(row0 + srow) * DD + schk * 8;
  if (tid < 64) {
    uint32_t lo = (uint32_t)(srow * 64 + schk * 16);
    uint64_t ga = (uint64_t)(uintptr_t)gsrc;
    asm volatile("global_load_async_to_lds_b128 %0, %1, off"
                 :: "v"(lo), "v"(ga), "v"(shesc)
                 : "memory");
  }

  v8f cr = {}; v8f ci = {};
  for (int k0 = 0; k0 < DD; k0 += 32) {
    asm volatile("s_wait_asynccnt 0x0" ::: "memory");
    __syncthreads();
    const int buf = (k0 >> 5) & 1;
    if (k0 + 32 < DD && tid < 64) {                   // stage next slab into buf^1
      uint32_t lo = (uint32_t)((buf ^ 1) * 1024 + srow * 64 + schk * 16);
      uint64_t ga = (uint64_t)(uintptr_t)(gsrc + k0 + 32);
      asm volatile("global_load_async_to_lds_b128 %0, %1, off"
                   :: "v"(lo), "v"(ga), "v"(shesc)
                   : "memory");
    }
    // A fragment from LDS via explicit ds_load_b128 pair (+ self-contained wait)
    Frag a;
    {
      const int m = lane & 15, half = lane >> 4;
      uint32_t laddr = (uint32_t)(buf * 1024 + m * 64 + 16 * half);
      v4u lo4, hi4;
      asm volatile("ds_load_b128 %0, %2\n\t"
                   "ds_load_b128 %1, %2 offset:32\n\t"
                   "s_wait_dscnt 0x0"
                   : "=v"(lo4), "=v"(hi4)
                   : "v"(laddr)
                   : "memory");
#pragma unroll
      for (int j = 0; j < 4; ++j) { a.u[j] = lo4[j]; a.u[4 + j] = hi4[j]; }
    }
    if (k0 + 32 < DD) {
      __builtin_prefetch(Wtr + (size_t)(col0 + (lane & 15)) * DD + k0 + 32, 0, 0);
      __builtin_prefetch(Wti + (size_t)(col0 + (lane & 15)) * DD + k0 + 32, 0, 0);
    }
    Frag br = load_b_nt(Wtr, DD, col0, k0, lane);
    Frag bi = load_b_nt(Wti, DD, col0, k0, lane);
    cr = wmma_bf(a, br, cr);
    ci = wmma_bf(a, bi, ci);
  }
  const int n = col0 + (lane & 15), half = lane >> 4;
#pragma unroll
  for (int r = 0; r < 8; ++r) {
    size_t idx = (size_t)(row0 + r + 8*half) * DD + n;
    outR[idx] = cr[r]; outI[idx] = ci[r];
  }
}

// flux_next = flux*fdecay + (x_cat @ Wi^T + bi)  -> ws + d_out tail
__global__ void k_flux(const float* __restrict__ xgr, const float* __restrict__ xgi,
                       const float* __restrict__ fr0, const float* __restrict__ fi0,
                       const float* __restrict__ Wi, const float* __restrict__ bi,
                       const float* __restrict__ dre, const float* __restrict__ dim_,
                       float* __restrict__ foutR, float* __restrict__ foutI,
                       float* __restrict__ out_flux) {
  int idx = blockIdx.x * blockDim.x + threadIdx.x;
  if (idx >= BB*DD) return;
  int b = idx / DD, j = idx % DD;
  float ar = bi[j], ai = bi[DD + j];
  const float* wr    = Wi + (size_t)j * 2*DD;
  const float* wiRow = Wi + (size_t)(DD + j) * 2*DD;
  for (int c = 0; c < DD; ++c) {
    float xr = xgr[b*DD + c], xi = xgi[b*DD + c];
    ar += xr * wr[c] + xi * wr[DD + c];
    ai += xr * wiRow[c] + xi * wiRow[DD + c];
  }
  float fdr = 1.0f / (1.0f + __expf(-dre[j]));
  float fdi = dim_[j];
  float fr = fr0[idx], fi = fi0[idx];
  float nr = fr * fdr - fi * fdi + ar;
  float ni = fr * fdi + fi * fdr + ai;
  foutR[idx] = nr; foutI[idx] = ni;
  out_flux[2*idx] = nr; out_flux[2*idx + 1] = ni;
}

// source = s_cat @ Wo^T + bo (complex), gate = sigmoid(s_cat @ Wg^T + bg)
__global__ void k_srcgate(const float* __restrict__ fR, const float* __restrict__ fI,
                          const float* __restrict__ Wo, const float* __restrict__ bo,
                          const float* __restrict__ Wg, const float* __restrict__ bg,
                          float* __restrict__ sR, float* __restrict__ sI, float* __restrict__ gate) {
  int idx = blockIdx.x * blockDim.x + threadIdx.x;
  if (idx >= BB*DD) return;
  int b = idx / DD, j = idx % DD;
  float ar = bo[j], ai = bo[DD + j], ag = bg[j];
  const float* wor = Wo + (size_t)j * 2*DD;
  const float* woi = Wo + (size_t)(DD + j) * 2*DD;
  const float* wg  = Wg + (size_t)j * 2*DD;
  for (int c = 0; c < DD; ++c) {
    float sr = fR[b*DD + c], si = fI[b*DD + c];
    ar += sr * wor[c] + si * wor[DD + c];
    ai += sr * woi[c] + si * woi[DD + c];
    ag += sr * wg[c]  + si * wg[DD + c];
  }
  sR[idx] = ar; sI[idx] = ai;
  gate[idx] = 1.0f / (1.0f + __expf(-ag));
}

// per-channel lam / op_decay / op_forcing / base_var vectors
__global__ void k_vec(const float* __restrict__ ld, const float* __restrict__ lf,
                      const float* __restrict__ lwr, const float* __restrict__ lwi,
                      const float* __restrict__ dt, const float* __restrict__ bnoise,
                      float* __restrict__ decR, float* __restrict__ decI,
                      float* __restrict__ forR, float* __restrict__ forI,
                      float* __restrict__ bvar) {
  int n = blockIdx.x * blockDim.x + threadIdx.x;
  if (n >= DD) return;
  float lre = tanhf(-__expf(ld[n]) + lwr[n]) * 0.3f;
  float lim = lf[n] + lwi[n];
  float dtr = dt[0];                    // DT_REF = 1.0
  float er = __expf(lre * dtr);
  float dr = er * __cosf(lim * dtr), di = er * __sinf(lim * dtr);
  decR[n] = dr; decI[n] = di;
  float nr = dr - 1.0f, ni = di;
  float der = lre + 1e-8f, dei = lim;
  float d2 = der * der + dei * dei;
  forR[n] = (nr * der + ni * dei) / d2;
  forI[n] = (ni * der - nr * dei) / d2;
  float bn = bnoise[0];
  float bv = bn * bn * expm1f(2.0f * lre * dtr) / (2.0f * lre + 1e-8f);
  bvar[n] = fmaxf(bv, 0.0f);
}

// h_pre = h_prev*op_decay + (xt*g + src*(1-g))*op_forcing ; bf16 copy of Re for MLP
__global__ void k_hpre(const float* __restrict__ xtR, const float* __restrict__ xtI,
                       const float* __restrict__ gate,
                       const float* __restrict__ sR, const float* __restrict__ sI,
                       const float* __restrict__ hpr, const float* __restrict__ hpi,
                       const float* __restrict__ decR, const float* __restrict__ decI,
                       const float* __restrict__ forR, const float* __restrict__ forI,
                       float* __restrict__ oR, float* __restrict__ oI,
                       uint16_t* __restrict__ hmbf) {
  size_t idx = (size_t)blockIdx.x * blockDim.x + threadIdx.x;
  if (idx >= (size_t)TT*DD) return;
  int t = (int)(idx / DD), n = (int)(idx % DD);
  int b = t >> 8;                        // SP = 256
  float g = gate[b*DD + n];
  float fr = xtR[idx] * g + sR[b*DD + n] * (1.0f - g);
  float fi = xtI[idx] * g + sI[b*DD + n] * (1.0f - g);
  float hr = hpr[idx], hi = hpi[idx];
  float dr = decR[n], di = decI[n], qr = forR[n], qi = forI[n];
  float nr = hr*dr - hi*di + fr*qr - fi*qi;
  float ni = hr*di + hi*dr + fr*qi + fi*qr;
  oR[idx] = nr; oI[idx] = ni; hmbf[idx] = f2bf(nr);
}

// t1 = silu(h_mean @ Wu1^T + bu1) -> bf16   (M=T, N=H, K=D, B is N x K "nt")
__global__ void k_mlp1(const uint16_t* __restrict__ A, const uint16_t* __restrict__ Bm,
                       const float* __restrict__ bias, uint16_t* __restrict__ out) {
  const int lane = threadIdx.x & 31;
  const int wid  = blockIdx.x * (blockDim.x >> 5) + (threadIdx.x >> 5);
  const int tn = wid & 15, tm = wid >> 4;            // (T/16)=256 x (H/16)=16
  const int row0 = tm * 16, col0 = tn * 16;
  v8f c = {};
  for (int k0 = 0; k0 < DD; k0 += 32) {
    Frag a = load_a_frag(A, DD, row0, k0, lane);
    Frag b = load_b_nt(Bm, DD, col0, k0, lane);
    c = wmma_bf(a, b, c);
  }
  const int n = col0 + (lane & 15), half = lane >> 4;
#pragma unroll
  for (int r = 0; r < 8; ++r) {
    float x = c[r] + bias[n];
    float s = x / (1.0f + __expf(-x));                 // silu
    out[(size_t)(row0 + r + 8*half) * HH + n] = f2bf(s);
  }
}

// u = softplus(t1 @ Wu2^T + bu2); std = sqrt(base_var*u); out = h_pre + noise*sqrt(.5)*std
__global__ void k_mlp2_final(const uint16_t* __restrict__ A, const uint16_t* __restrict__ Bm,
                             const float* __restrict__ bias, const float* __restrict__ bvar,
                             const float* __restrict__ hR, const float* __restrict__ hI,
                             const float* __restrict__ nR, const float* __restrict__ nI,
                             float* __restrict__ out) {
  const int lane = threadIdx.x & 31;
  const int wid  = blockIdx.x * (blockDim.x >> 5) + (threadIdx.x >> 5);
  const int tn = wid & 63, tm = wid >> 6;            // (T/16)=256 x (D/16)=64
  const int row0 = tm * 16, col0 = tn * 16;
  v8f c = {};
  for (int k0 = 0; k0 < HH; k0 += 32) {
    Frag a = load_a_frag(A, HH, row0, k0, lane);
    Frag b = load_b_nt(Bm, HH, col0, k0, lane);
    c = wmma_bf(a, b, c);
  }
  const int n = col0 + (lane & 15), half = lane >> 4;
#pragma unroll
  for (int r = 0; r < 8; ++r) {
    size_t idx = (size_t)(row0 + r + 8*half) * DD + n;
    float x = c[r] + bias[n];
    float u = (x > 20.0f) ? x : log1pf(__expf(x));     // softplus
    float sd = sqrtf(bvar[n] * u) * 0.70710678f;       // sqrt(0.5)*std
    out[2*idx]     = hR[idx] + nR[idx] * sd;
    out[2*idx + 1] = hI[idx] + nI[idx] * sd;
  }
}

// ---------------- host launcher ----------------

extern "C" void kernel_launch(void* const* d_in, const int* in_sizes, int n_in,
                              void* d_out, int out_size, void* d_ws, size_t ws_size,
                              hipStream_t stream) {
  (void)in_sizes; (void)n_in; (void)out_size; (void)ws_size;
  const float* x_input  = (const float*)d_in[0];
  const float* hprev_re = (const float*)d_in[1];
  const float* hprev_im = (const float*)d_in[2];
  const float* xg_re    = (const float*)d_in[3];
  const float* xg_im    = (const float*)d_in[4];
  const float* flux_re  = (const float*)d_in[5];
  const float* flux_im  = (const float*)d_in[6];
  const float* dt       = (const float*)d_in[7];
  const float* v_raw_re = (const float*)d_in[10];
  const float* v_raw_im = (const float*)d_in[11];
  const float* log_sig  = (const float*)d_in[12];
  const float* dft_w    = (const float*)d_in[13];
  const float* decay_re = (const float*)d_in[14];
  const float* decay_im = (const float*)d_in[15];
  const float* Wi       = (const float*)d_in[16];
  const float* bi       = (const float*)d_in[17];
  const float* Wo       = (const float*)d_in[18];
  const float* bo       = (const float*)d_in[19];
  const float* Wg       = (const float*)d_in[20];
  const float* bg       = (const float*)d_in[21];
  const float* ld       = (const float*)d_in[22];
  const float* lf       = (const float*)d_in[23];
  const float* law_re   = (const float*)d_in[24];
  const float* law_im   = (const float*)d_in[25];
  const float* bnoise   = (const float*)d_in[26];
  const float* Wu1      = (const float*)d_in[27];
  const float* bu1      = (const float*)d_in[28];
  const float* Wu2      = (const float*)d_in[29];
  const float* bu2      = (const float*)d_in[30];
  const float* noise_re = (const float*)d_in[31];
  const float* noise_im = (const float*)d_in[32];
  float* out = (float*)d_out;

  char* ws = (char*)d_ws;
  size_t off = 0;
  auto take = [&](size_t bytes) -> void* {
    void* p = ws + off;
    off = (off + bytes + 255) & ~(size_t)255;
    return p;
  };
  const size_t D2 = (size_t)DD*DD, TD = (size_t)TT*DD, TH = (size_t)TT*HH, BD = (size_t)BB*DD;

  uint16_t* Pre  = (uint16_t*)take(D2*2);   // P (natural, A operand)
  uint16_t* Pim  = (uint16_t*)take(D2*2);
  uint16_t* PtR  = (uint16_t*)take(D2*2);   // P^T (first B operand)
  uint16_t* PtI  = (uint16_t*)take(D2*2);
  uint16_t* MTaR = (uint16_t*)take(D2*2);   // M^T double buffer
  uint16_t* MTaI = (uint16_t*)take(D2*2);
  uint16_t* MTbR = (uint16_t*)take(D2*2);
  uint16_t* MTbI = (uint16_t*)take(D2*2);
  uint16_t* Wtr  = (uint16_t*)take(D2*2);
  uint16_t* Wti  = (uint16_t*)take(D2*2);
  uint16_t* xbf  = (uint16_t*)take(TD*2);
  uint16_t* wu1b = (uint16_t*)take((size_t)HH*DD*2);
  uint16_t* wu2b = (uint16_t*)take((size_t)DD*HH*2);
  uint16_t* t1bf = (uint16_t*)take(TH*2);
  uint16_t* hmbf = (uint16_t*)take(TD*2);
  float* Vre  = (float*)take(D2*4);
  float* Vim  = (float*)take(D2*4);
  float* xtR  = (float*)take(TD*4);
  float* xtI  = (float*)take(TD*4);
  float* hpR  = (float*)take(TD*4);
  float* hpI  = (float*)take(TD*4);
  float* flR  = (float*)take(BD*4);
  float* flI  = (float*)take(BD*4);
  float* srR  = (float*)take(BD*4);
  float* srI  = (float*)take(BD*4);
  float* gate = (float*)take(BD*4);
  float* decR = (float*)take(DD*4);
  float* decI = (float*)take(DD*4);
  float* forR = (float*)take(DD*4);
  float* forI = (float*)take(DD*4);
  float* bvar = (float*)take(DD*4);

  // bf16 conversions of GEMM operands
  k_f32_to_bf16<<<(int)(TD/256), 256, 0, stream>>>(x_input, xbf, (int)TD);
  k_f32_to_bf16<<<(int)((size_t)HH*DD/256), 256, 0, stream>>>(Wu1, wu1b, HH*DD);
  k_f32_to_bf16<<<(int)((size_t)DD*HH/256), 256, 0, stream>>>(Wu2, wu2b, DD*HH);

  // Cayley via Neumann series on P = A^T  (Vt accumulates V^T)
  k_prep<<<(int)(D2/256), 256, 0, stream>>>(v_raw_re, v_raw_im, Pre, Pim, PtR, PtI, Vre, Vim);
  {
    const uint16_t *cR = PtR, *cI = PtI;
    for (int k = 2; k <= NTERMS; ++k) {
      uint16_t* oR = (k & 1) ? MTbR : MTaR;
      uint16_t* oI = (k & 1) ? MTbI : MTaI;
      float sign2 = (k & 1) ? -2.0f : 2.0f;
      k_neumann<<<1024, 128, 0, stream>>>(Pre, Pim, cR, cI, oR, oI, Vre, Vim, sign2);
      cR = oR; cI = oI;
    }
  }
  k_combine<<<(int)(D2/256), 256, 0, stream>>>(Vre, Vim, log_sig, dft_w, Wtr, Wti);

  // independent small pipelines
  k_vec<<<DD/256, 256, 0, stream>>>(ld, lf, law_re, law_im, dt, bnoise, decR, decI, forR, forI, bvar);
  k_flux<<<(int)(BD/256), 256, 0, stream>>>(xg_re, xg_im, flux_re, flux_im, Wi, bi,
                                            decay_re, decay_im, flR, flI, out + 2*TD);
  k_srcgate<<<(int)(BD/256), 256, 0, stream>>>(flR, flI, Wo, bo, Wg, bg, srR, srI, gate);

  // x_tilde big GEMM (WMMA + async global->LDS A staging + ds_load fragments)
  k_xtilde<<<4096, 128, 0, stream>>>(xbf, Wtr, Wti, xtR, xtI);

  // h_pre elementwise
  k_hpre<<<(int)(TD/256), 256, 0, stream>>>(xtR, xtI, gate, srR, srI, hprev_re, hprev_im,
                                            decR, decI, forR, forI, hpR, hpI, hmbf);

  // u-MLP (WMMA) with fused epilogues; final writes interleaved complex h_next
  k_mlp1<<<1024, 128, 0, stream>>>(hmbf, wu1b, bu1, t1bf);
  k_mlp2_final<<<4096, 128, 0, stream>>>(t1bf, wu2b, bu2, bvar, hpR, hpI, noise_re, noise_im, out);
}
